// PCD_Align_41094247088686
// MI455X (gfx1250) — matrix operand order repaired and checked by reading, not compiled, not run on hardware
//
#include <hip/hip_runtime.h>

typedef unsigned short u16;
typedef unsigned int   u32;
typedef unsigned long long u64;
typedef __attribute__((ext_vector_type(16))) __bf16 v16bf;
typedef __attribute__((ext_vector_type(8)))  float  v8f;

__device__ __forceinline__ u16 f2bf(float f) {
  u32 u = __builtin_bit_cast(u32, f);
  u32 r = (u + 0x7FFFu + ((u >> 16) & 1u)) >> 16;  // RNE
  return (u16)r;
}
__device__ __forceinline__ float bf2f(u16 h) {
  u32 u = ((u32)h) << 16;
  return __builtin_bit_cast(float, u);
}

union FragB { v16bf v; uint4 q[2]; };

// ---------------------------------------------------------------------------
// NCHW fp32 -> NHWC bf16 (C = 128)
// ---------------------------------------------------------------------------
__global__ void __launch_bounds__(256)
nchw2nhwc_bf16(const float* __restrict__ in, u16* __restrict__ out,
               int Bn, int Hh, int Ww) {
  size_t total = (size_t)Bn * 128 * Hh * Ww;
  for (size_t i = (size_t)blockIdx.x * 256 + threadIdx.x; i < total;
       i += (size_t)gridDim.x * 256) {
    int w = (int)(i % Ww);
    size_t t = i / Ww;
    int h = (int)(t % Hh); t /= Hh;
    int c = (int)(t % 128);
    int b = (int)(t / 128);
    out[((size_t)(b * Hh + h) * Ww + w) * 128 + c] = f2bf(in[i]);
  }
}

// ---------------------------------------------------------------------------
// Pack OIHW fp32 weights -> [tap][cout_pad][cin] bf16, pad bias to cout_pad.
// ---------------------------------------------------------------------------
__global__ void __launch_bounds__(256)
pack_w_kernel(const float* __restrict__ w, const float* __restrict__ bsrc,
              u16* __restrict__ wout, float* __restrict__ bout,
              int cout, int cin, int cop) {
  size_t tid0 = (size_t)blockIdx.x * 256 + threadIdx.x;
  if (tid0 < (size_t)cop) bout[tid0] = (tid0 < (size_t)cout) ? bsrc[tid0] : 0.f;
  size_t total = (size_t)9 * cop * cin;
  for (size_t i = tid0; i < total; i += (size_t)gridDim.x * 256) {
    int ci = (int)(i % cin);
    size_t t = i / cin;
    int co = (int)(t % cop);
    int k  = (int)(t / cop);
    float v = 0.f;
    if (co < cout) {
      int ky = k / 3, kx = k % 3;
      v = w[(((size_t)co * cin + ci) * 3 + ky) * 3 + kx];
    }
    wout[((size_t)k * cop + co) * cin + ci] = f2bf(v);
  }
}

// ---------------------------------------------------------------------------
// Bilinear x2 upsample, NHWC bf16 C=128, half-pixel centers, edge clamp,
// fused output scale (offsets get *2).
// ---------------------------------------------------------------------------
__global__ void __launch_bounds__(256)
up2_bilinear(const u16* __restrict__ in, u16* __restrict__ out,
             int Bn, int Hi, int Wi, float scale) {
  int Ho = Hi * 2, Wo = Wi * 2;
  size_t total = (size_t)Bn * Ho * Wo * 128;
  for (size_t i = (size_t)blockIdx.x * 256 + threadIdx.x; i < total;
       i += (size_t)gridDim.x * 256) {
    int c = (int)(i & 127);
    size_t p = i >> 7;
    int xo = (int)(p % Wo); p /= Wo;
    int yo = (int)(p % Ho);
    int bi = (int)(p / Ho);
    float sy = (yo + 0.5f) * 0.5f - 0.5f;
    float sx = (xo + 0.5f) * 0.5f - 0.5f;
    float yf = floorf(sy), xf = floorf(sx);
    float wy = sy - yf, wx = sx - xf;
    int y0 = (int)yf, x0 = (int)xf;
    int y1 = y0 + 1, x1 = x0 + 1;
    y0 = y0 < 0 ? 0 : (y0 > Hi - 1 ? Hi - 1 : y0);
    y1 = y1 < 0 ? 0 : (y1 > Hi - 1 ? Hi - 1 : y1);
    x0 = x0 < 0 ? 0 : (x0 > Wi - 1 ? Wi - 1 : x0);
    x1 = x1 < 0 ? 0 : (x1 > Wi - 1 ? Wi - 1 : x1);
    const u16* bp = in + (size_t)bi * Hi * Wi * 128 + c;
    float v = (1.f - wy) * (1.f - wx) * bf2f(bp[((size_t)y0 * Wi + x0) * 128])
            + (1.f - wy) * wx         * bf2f(bp[((size_t)y0 * Wi + x1) * 128])
            + wy * (1.f - wx)         * bf2f(bp[((size_t)y1 * Wi + x0) * 128])
            + wy * wx                 * bf2f(bp[((size_t)y1 * Wi + x1) * 128]);
    out[i] = f2bf(v * scale);
  }
}

// ---------------------------------------------------------------------------
// Implicit-GEMM 3x3 conv (stride 1, pad 1) on WMMA bf16.
//   in0/in1: NHWC bf16, 128 ch each (in1 != null => channel-concat, cin=256)
//   wgt:     [9][cop][cin] bf16, bias: [cop] f32
//   out:     NHWC bf16 (out_bf) or NHWC f32 (out_f), channel stride = cop
// Block: 256 thr = 8 waves (4 along M x 2 along N); tile = 64 px x 128 cout;
// each wave owns 16 px x 64 cout (4 accumulators) so every A fragment read
// from LDS feeds 4 WMMAs. A-slab staged with async direct-to-LDS copies.
// ---------------------------------------------------------------------------
__global__ void __launch_bounds__(256)
conv3x3_wmma(const u16* __restrict__ in0, const u16* __restrict__ in1, int cin,
             const u16* __restrict__ wgt, const float* __restrict__ bias, int cop,
             u16* __restrict__ out_bf, float* __restrict__ out_f, int fuse_lrelu,
             int Bn, int Hh, int Ww, int ntx) {
  __shared__ u16 As[3][66][32];  // 3 rows x (64+2 halo) x 32 channels @ LDS base
  const int tid  = threadIdx.x;
  const int lane = tid & 31;
  const int wave = tid >> 5;
  const int mw = wave & 3;   // M wave (16 px)
  const int nw = wave >> 2;  // N wave (64 cout)
  const int tx = blockIdx.x % ntx;
  const int rb = blockIdx.x / ntx;
  const int y  = rb % Hh;
  const int bi = rb / Hh;
  const int x0 = tx * 64;
  const int coBase = blockIdx.y * 128 + nw * 64;

  v8f cacc[4] = {v8f{}, v8f{}, v8f{}, v8f{}};
  const int nch = cin >> 5;
  const int lm = lane & 15;
  const int lg = lane >> 4;

  for (int ch = 0; ch < nch; ++ch) {
    const u16* src = (ch < 4) ? in0 : in1;
    const int cch = (ch & 3) * 32;

    // prefetch next ci-chunk of this wave's weight rows into GL2
    if (ch + 1 < nch) {
      const u16* pf = wgt + ((size_t)0 * cop + (coBase + lm)) * cin + (ch + 1) * 32;
      __builtin_prefetch(pf, 0, 1);
    }

    // ---- stage A slab (3 rows x 66 px x 32 ch) via async global->LDS DMA ----
    for (int idx = tid; idx < 3 * 66; idx += 256) {
      int r  = idx / 66;
      int xl = idx % 66;
      int yg = y + r - 1;
      int xg = x0 + xl - 1;
      u32 lofs = (u32)(((r * 66 + xl) * 32) * 2);  // byte offset of As[r][xl][0]
      if (yg >= 0 && yg < Hh && xg >= 0 && xg < Ww) {
        u64 ga = (u64)(const void*)(src + ((size_t)(bi * Hh + yg) * Ww + xg) * 128 + cch);
        // IOFFSET is added to both the LDS and the global address (ISA 10.x),
        // so one address pair covers the 64B slot.
        asm volatile(
            "global_load_async_to_lds_b128 %0, %1, off\n\t"
            "global_load_async_to_lds_b128 %0, %1, off offset:16\n\t"
            "global_load_async_to_lds_b128 %0, %1, off offset:32\n\t"
            "global_load_async_to_lds_b128 %0, %1, off offset:48"
            :: "v"(lofs), "v"(ga) : "memory");
      } else {
        uint4 z = make_uint4(0u, 0u, 0u, 0u);
        uint4* q = (uint4*)&As[r][xl][0];
        q[0] = z; q[1] = z; q[2] = z; q[3] = z;
      }
    }
    asm volatile("s_wait_asynccnt 0x0" ::: "memory");
    __syncthreads();

    const int px = mw * 16 + lm;
#pragma unroll
    for (int k = 0; k < 9; ++k) {
      const int ky = k / 3, kx = k % 3;
      FragB af;
      // A fragment per ISA 16-bit layout: lanes0-15 K g*8+0..7 / 16+g*8..
      const u16* ap = &As[ky][px + kx][0];
      af.q[0] = *(const uint4*)(ap + lg * 8);
      af.q[1] = *(const uint4*)(ap + 16 + lg * 8);
#pragma unroll
      for (int f = 0; f < 4; ++f) {
        FragB bf;
        const u16* wp = wgt + ((size_t)k * cop + (coBase + f * 16 + lm)) * cin
                        + ch * 32 + lg * 8;
        bf.q[0] = *(const uint4*)(wp);
        bf.q[1] = *(const uint4*)(wp + 16);
        cacc[f] = __builtin_amdgcn_wmma_f32_16x16x32_bf16(
            false, af.v, false, bf.v, (short)0, cacc[f], false, false);
      }
    }
    __syncthreads();
  }

  // ---- epilogue: bias + optional lrelu, NHWC store ----
#pragma unroll
  for (int f = 0; f < 4; ++f) {
    const int co = coBase + f * 16 + lm;
    const float bv = bias[co];
#pragma unroll
    for (int v = 0; v < 8; ++v) {
      const int m = v + 8 * lg;
      const int x = x0 + mw * 16 + m;
      if (x < Ww) {
        float val = cacc[f][v] + bv;
        if (fuse_lrelu) val = (val >= 0.f) ? val : 0.1f * val;
        const size_t pix = (size_t)(bi * Hh + y) * Ww + x;
        if (out_f) out_f[pix * cop + co] = val;
        else       out_bf[pix * cop + co] = f2bf(val);
      }
    }
  }
}

// ---------------------------------------------------------------------------
// Modulated deformable 3x3 conv (DG=8 groups) fused with WMMA GEMM.
//   x:   NHWC bf16 128ch;  om: NHWC f32 256ch (o_y[72] | o_x[72] | mask[72])
//   wgt: [9][128][128] bf16;  bias: [128] f32
//   out: NHWC bf16, or fp32 NCHW (final cascade output into d_out)
// Block: 256 thr = 8 waves; tile = 16 px x 128 cout (each wave 16 couts).
// ---------------------------------------------------------------------------
__global__ void __launch_bounds__(256)
dcn_wmma(const u16* __restrict__ x, const float* __restrict__ om,
         const u16* __restrict__ wgt, const float* __restrict__ bias,
         u16* __restrict__ out_bf, float* __restrict__ out_f_nchw,
         int fuse_lrelu, int Bn, int Hh, int Ww) {
  __shared__ u16 samp[16][128];
  const int tid  = threadIdx.x;
  const int lane = tid & 31;
  const int wave = tid >> 5;
  const int ntx = Ww >> 4;
  const int tx = blockIdx.x % ntx;
  const int rb = blockIdx.x / ntx;
  const int y  = rb % Hh;
  const int bi = rb / Hh;
  const int x0 = tx * 16;

  // sampling assignment: thread -> (pixel, 8 channels)
  const int spx  = tid >> 4;
  const int sc   = (tid & 15) * 8;
  const int gidx = sc >> 4;          // deformable group of these 8 channels
  const int xs   = x0 + spx;
  const size_t ombase = ((size_t)(bi * Hh + y) * Ww + xs) * 256;

  const int lm = lane & 15;
  const int lg = lane >> 4;
  const int co = wave * 16 + lm;
  v8f acc = {};

  for (int k = 0; k < 9; ++k) {
    const int ky = k / 3 - 1, kx = k % 3 - 1;
    const float oy = om[ombase + gidx * 9 + k];
    const float ox = om[ombase + 72 + gidx * 9 + k];
    const float mv = om[ombase + 144 + gidx * 9 + k];
    const float msk = 1.f / (1.f + expf(-mv));
    const float py  = oy + (float)y + (float)ky;
    const float pxs = ox + (float)xs + (float)kx;
    const float yf = floorf(py), xf = floorf(pxs);
    const float wy = py - yf, wx = pxs - xf;
    const int iy0 = (int)yf, ix0 = (int)xf;
    float vals[8];
#pragma unroll
    for (int j = 0; j < 8; ++j) vals[j] = 0.f;
    const float wq[4] = {(1.f - wy) * (1.f - wx), (1.f - wy) * wx,
                         wy * (1.f - wx),         wy * wx};
#pragma unroll
    for (int t = 0; t < 4; ++t) {
      const int yy = iy0 + (t >> 1), xx = ix0 + (t & 1);
      if (yy >= 0 && yy < Hh && xx >= 0 && xx < Ww) {
        const uint4 q = *(const uint4*)(x +
            ((size_t)(bi * Hh + yy) * Ww + xx) * 128 + sc);
        const u16* hv = (const u16*)&q;
#pragma unroll
        for (int j = 0; j < 8; ++j) vals[j] += wq[t] * bf2f(hv[j]);
      }
    }
    union { uint4 q; u16 h[8]; } pk;
#pragma unroll
    for (int j = 0; j < 8; ++j) pk.h[j] = f2bf(vals[j] * msk);
    *(uint4*)&samp[spx][sc] = pk.q;
    __syncthreads();

#pragma unroll
    for (int ch = 0; ch < 4; ++ch) {
      FragB af, bf;
      const u16* ap = &samp[lm][ch * 32];
      af.q[0] = *(const uint4*)(ap + lg * 8);
      af.q[1] = *(const uint4*)(ap + 16 + lg * 8);
      const u16* wp = wgt + ((size_t)k * 128 + co) * 128 + ch * 32 + lg * 8;
      bf.q[0] = *(const uint4*)(wp);
      bf.q[1] = *(const uint4*)(wp + 16);
      acc = __builtin_amdgcn_wmma_f32_16x16x32_bf16(false, af.v, false, bf.v,
                                                    (short)0, acc, false, false);
    }
    __syncthreads();
  }

  const float bv = bias[co];
#pragma unroll
  for (int v = 0; v < 8; ++v) {
    const int m = v + 8 * lg;
    const int xg = x0 + m;
    float val = acc[v] + bv;
    if (fuse_lrelu) val = (val >= 0.f) ? val : 0.1f * val;
    if (out_f_nchw)
      out_f_nchw[(((size_t)bi * 128 + co) * Hh + y) * Ww + xg] = val;
    else
      out_bf[((size_t)(bi * Hh + y) * Ww + xg) * 128 + co] = f2bf(val);
  }
}

// ---------------------------------------------------------------------------
// Host orchestration
// ---------------------------------------------------------------------------
struct PackedConv { u16* w; float* b; int cin; int cop; };

extern "C" void kernel_launch(void* const* d_in, const int* in_sizes, int n_in,
                              void* d_out, int out_size, void* d_ws, size_t ws_size,
                              hipStream_t stream) {
  (void)in_sizes; (void)n_in; (void)out_size; (void)ws_size;
  const int Bn = 2;
  const int H1 = 128, W1 = 128, H2 = 64, W2 = 64, H3 = 32, W3 = 32;

  char* wsp = (char*)d_ws;
  size_t off = 0;
  auto alloc = [&](size_t bytes) -> void* {
    void* p = wsp + off;
    off += (bytes + 255) & ~(size_t)255;
    return p;
  };

  const size_t S1 = (size_t)Bn * H1 * W1 * 128 * sizeof(u16);
  const size_t S2 = (size_t)Bn * H2 * W2 * 128 * sizeof(u16);
  const size_t S3 = (size_t)Bn * H3 * W3 * 128 * sizeof(u16);
  const size_t OM = (size_t)Bn * H1 * W1 * 256 * sizeof(float);

  // ---- pack all 20 conv weights ----
  auto pack = [&](int pidx, int cin, int cop) -> PackedConv {
    PackedConv pc;
    pc.w = (u16*)alloc((size_t)9 * cop * cin * sizeof(u16));
    pc.b = (float*)alloc((size_t)cop * sizeof(float));
    pc.cin = cin; pc.cop = cop;
    const float* wsrc = (const float*)d_in[6 + 2 * pidx];
    const float* bsrc = (const float*)d_in[6 + 2 * pidx + 1];
    int cout = (cop == 256) ? 216 : 128;  // om convs output DG*27=216
    pack_w_kernel<<<dim3(256), dim3(256), 0, stream>>>(wsrc, bsrc, pc.w, pc.b,
                                                       cout, cin, cop);
    return pc;
  };
  PackedConv P[20];
  P[0]  = pack(0, 256, 128);  // L3_oc1
  P[1]  = pack(1, 128, 128);  // L3_oc2
  P[2]  = pack(2, 128, 256);  // L3_om
  P[3]  = pack(3, 128, 128);  // L3_dcn
  P[4]  = pack(4, 256, 128);  // L2_oc1
  P[5]  = pack(5, 256, 128);  // L2_oc2
  P[6]  = pack(6, 128, 128);  // L2_oc3
  P[7]  = pack(7, 128, 256);  // L2_om
  P[8]  = pack(8, 128, 128);  // L2_dcn
  P[9]  = pack(9, 256, 128);  // L2_fc
  P[10] = pack(10, 256, 128); // L1_oc1
  P[11] = pack(11, 256, 128); // L1_oc2
  P[12] = pack(12, 128, 128); // L1_oc3
  P[13] = pack(13, 128, 256); // L1_om
  P[14] = pack(14, 128, 128); // L1_dcn
  P[15] = pack(15, 256, 128); // L1_fc
  P[16] = pack(16, 256, 128); // cas_oc1
  P[17] = pack(17, 128, 128); // cas_oc2
  P[18] = pack(18, 128, 256); // cas_om
  P[19] = pack(19, 128, 128); // cas_dcn

  // ---- convert inputs to NHWC bf16 ----
  u16* nb1 = (u16*)alloc(S1); u16* nb2 = (u16*)alloc(S2); u16* nb3 = (u16*)alloc(S3);
  u16* rf1 = (u16*)alloc(S1); u16* rf2 = (u16*)alloc(S2); u16* rf3 = (u16*)alloc(S3);
  auto cvt = [&](int i, u16* dst, int Hh, int Ww) {
    nchw2nhwc_bf16<<<dim3(2048), dim3(256), 0, stream>>>(
        (const float*)d_in[i], dst, Bn, Hh, Ww);
  };
  cvt(0, nb1, H1, W1); cvt(1, nb2, H2, W2); cvt(2, nb3, H3, W3);
  cvt(3, rf1, H1, W1); cvt(4, rf2, H2, W2); cvt(5, rf3, H3, W3);

  float* om = (float*)alloc(OM);  // reused at every level
  u16* t3a = (u16*)alloc(S3); u16* t3b = (u16*)alloc(S3); u16* fea3 = (u16*)alloc(S3);
  u16* uoff3 = (u16*)alloc(S2); u16* ufea3 = (u16*)alloc(S2);
  u16* t2a = (u16*)alloc(S2); u16* t2b = (u16*)alloc(S2); u16* t2c = (u16*)alloc(S2);
  u16* d2  = (u16*)alloc(S2); u16* fea2 = (u16*)alloc(S2);
  u16* uoff2 = (u16*)alloc(S1); u16* ufea2 = (u16*)alloc(S1);
  u16* t1a = (u16*)alloc(S1); u16* t1b = (u16*)alloc(S1); u16* t1c = (u16*)alloc(S1);
  u16* d1  = (u16*)alloc(S1); u16* fea1 = (u16*)alloc(S1);
  u16* c1b = (u16*)alloc(S1); u16* c2b = (u16*)alloc(S1);

  auto conv = [&](const u16* a, const u16* b, const PackedConv& pc,
                  u16* obf, float* of, int lrelu, int Hh, int Ww) {
    int ntx = (Ww + 63) / 64;
    dim3 grid((unsigned)(ntx * Hh * Bn), (unsigned)(pc.cop >> 7));
    conv3x3_wmma<<<grid, dim3(256), 0, stream>>>(
        a, b, b ? 256 : 128, pc.w, pc.b, pc.cop, obf, of, lrelu, Bn, Hh, Ww, ntx);
  };
  auto dcn = [&](const u16* xx, const PackedConv& pc,
                 u16* obf, float* of, int lrelu, int Hh, int Ww) {
    dim3 grid((unsigned)((Ww / 16) * Hh * Bn));
    dcn_wmma<<<grid, dim3(256), 0, stream>>>(xx, om, pc.w, pc.b, obf, of,
                                             lrelu, Bn, Hh, Ww);
  };
  auto up = [&](const u16* src, u16* dst, int Hi, int Wi, float scale) {
    up2_bilinear<<<dim3(2048), dim3(256), 0, stream>>>(src, dst, Bn, Hi, Wi, scale);
  };

  // ---- L3 ----
  conv(nb3, rf3, P[0], t3a, nullptr, 1, H3, W3);
  conv(t3a, nullptr, P[1], t3b, nullptr, 1, H3, W3);           // L3_off
  conv(t3b, nullptr, P[2], nullptr, om, 0, H3, W3);            // offset+mask
  dcn(nb3, P[3], fea3, nullptr, 1, H3, W3);                    // L3_fea (lrelu)
  up(t3b, uoff3, H3, W3, 2.f);
  up(fea3, ufea3, H3, W3, 1.f);
  // ---- L2 ----
  conv(nb2, rf2, P[4], t2a, nullptr, 1, H2, W2);
  conv(t2a, uoff3, P[5], t2b, nullptr, 1, H2, W2);
  conv(t2b, nullptr, P[6], t2c, nullptr, 1, H2, W2);           // L2_off
  conv(t2c, nullptr, P[7], nullptr, om, 0, H2, W2);
  dcn(nb2, P[8], d2, nullptr, 0, H2, W2);                      // no lrelu
  conv(d2, ufea3, P[9], fea2, nullptr, 1, H2, W2);             // L2_fea
  up(t2c, uoff2, H2, W2, 2.f);
  up(fea2, ufea2, H2, W2, 1.f);
  // ---- L1 ----
  conv(nb1, rf1, P[10], t1a, nullptr, 1, H1, W1);
  conv(t1a, uoff2, P[11], t1b, nullptr, 1, H1, W1);
  conv(t1b, nullptr, P[12], t1c, nullptr, 1, H1, W1);          // L1_off
  conv(t1c, nullptr, P[13], nullptr, om, 0, H1, W1);
  dcn(nb1, P[14], d1, nullptr, 0, H1, W1);                     // no lrelu
  conv(d1, ufea2, P[15], fea1, nullptr, 0, H1, W1);            // L1_fea (no lrelu)
  // ---- cascade ----
  conv(fea1, rf1, P[16], c1b, nullptr, 1, H1, W1);
  conv(c1b, nullptr, P[17], c2b, nullptr, 1, H1, W1);
  conv(c2b, nullptr, P[18], nullptr, om, 0, H1, W1);
  dcn(fea1, P[19], nullptr, (float*)d_out, 1, H1, W1);         // f32 NCHW + lrelu
}
